// MotionTransformer_27169963114866
// MI455X (gfx1250) — matrix operand order repaired and checked by reading, not compiled, not run on hardware
//
#include <hip/hip_runtime.h>
#include <hip/hip_bf16.h>
#include <math.h>

typedef __hip_bfloat16 bf16;
typedef __attribute__((ext_vector_type(16))) __bf16 v16bf;
typedef __attribute__((ext_vector_type(8)))  float  v8f;
typedef __attribute__((ext_vector_type(4)))  unsigned int u32x4;

#define B_    2048
#define T_    20
#define S_    21
#define DIN_  66
#define D_    512
#define FF_   1024
#define H_    8
#define DH_   64
#define ROWS  (B_*S_)   // 43008
#define XROWS (B_*T_)   // 40960

// ---------------------------------------------------------------------------
// Async global -> LDS copy (ASYNCcnt-tracked). IOFFSET applies to BOTH the LDS
// destination and the global source (ISA 15.18.3), so a 32B row segment is two
// issues from one address pair. `base` is a dummy escaping pointer so the
// compiler knows the asm writes the shared array (keeps the ds_loads alive).
// ---------------------------------------------------------------------------
__device__ __forceinline__ void async_b128(unsigned lds, const void* g, void* base) {
  asm volatile("global_load_async_to_lds_b128 %0, %1, off"
               :: "v"(lds), "v"(g), "v"(base) : "memory");
}
__device__ __forceinline__ void async_b128_o16(unsigned lds, const void* g, void* base) {
  asm volatile("global_load_async_to_lds_b128 %0, %1, off offset:16"
               :: "v"(lds), "v"(g), "v"(base) : "memory");
}
__device__ __forceinline__ void async_wait0() {
  asm volatile("s_wait_asynccnt 0x0" ::: "memory");
}

// ---------------------------------------------------------------------------
// Tiled bf16 GEMM with WMMA: C = epilogue(A[MxK] * B[KxN])
// A: bf16 row-major [M][K] (K padded to mult of 32)
// Bw: bf16 [N][K] (pre-transposed weights)
// Tiles: 128x128x32, 8 waves, each wave computes 64x32 (4x2 16x16 tiles).
// Staging: double-buffered async global->LDS (no VGPR round trip).
// ---------------------------------------------------------------------------
union Frag { u32x4 u[2]; v16bf v; };

// single LDS allocation => base offset 0; layout: sA[2][4096] | sB[2][4096] halfs
#define SA_IDX(buf, idx) (((buf) << 12) + (idx))
#define SB_IDX(buf, idx) (8192 + ((buf) << 12) + (idx))

__global__ __launch_bounds__(256) void gemm_bf16_kernel(
    const bf16* __restrict__ A, const bf16* __restrict__ Bw,
    const float* __restrict__ bias, const float* __restrict__ res,
    float* __restrict__ outF, bf16* __restrict__ outBf,
    int M, int N, int K, int act)
{
  __shared__ unsigned short smem[4 * 4096];   // 32 KB

  const int tid  = threadIdx.x;
  const int lane = tid & 31;
  const int wave = tid >> 5;
  const int wm   = wave >> 2;   // 0..1
  const int wn   = wave & 3;    // 0..3
  const int l16  = lane & 15;
  const bool hi  = lane >= 16;

  const long blockRow = (long)blockIdx.x * 128;
  const long blockCol = (long)blockIdx.y * 128;

  // Loader: each of 256 threads moves 32B of A and 32B of B per k-stage.
  const int  ldr  = tid >> 1;      // 0..127 (row in tile)
  const int  lseg = tid & 1;       // which 16-half segment
  const bf16* aG = A  + (blockRow + ldr) * (long)K + lseg * 16;
  const bf16* bG = Bw + (blockCol + ldr) * (long)K + lseg * 16;
  const int ldsOff = ldr * 32 + lseg * 16;   // halfs within tile

  v8f acc[4][2];
  {
    v8f z = {0.f,0.f,0.f,0.f,0.f,0.f,0.f,0.f};
    #pragma unroll
    for (int i = 0; i < 4; ++i)
      #pragma unroll
      for (int j = 0; j < 2; ++j) acc[i][j] = z;
  }

  const int nk = K >> 5;

  // stage 0: async-stage k-tile 0 into buffer 0
  async_b128    (2u * SA_IDX(0, ldsOff), aG, smem);
  async_b128_o16(2u * SA_IDX(0, ldsOff), aG, smem);
  async_b128    (2u * SB_IDX(0, ldsOff), bG, smem);
  async_b128_o16(2u * SB_IDX(0, ldsOff), bG, smem);
  async_wait0();
  __syncthreads();

  for (int kt = 0; kt < nk; ++kt) {
    const int buf  = kt & 1;
    const bool more = (kt + 1) < nk;
    if (more) {
      const bf16* ap = aG + (long)(kt + 1) * 32;
      const bf16* bp = bG + (long)(kt + 1) * 32;
      async_b128    (2u * SA_IDX(buf ^ 1, ldsOff), ap, smem);
      async_b128_o16(2u * SA_IDX(buf ^ 1, ldsOff), ap, smem);
      async_b128    (2u * SB_IDX(buf ^ 1, ldsOff), bp, smem);
      async_b128_o16(2u * SB_IDX(buf ^ 1, ldsOff), bp, smem);
      if (kt + 2 < nk) {
        __builtin_prefetch(aG + (long)(kt + 2) * 32, 0, 3);
        __builtin_prefetch(bG + (long)(kt + 2) * 32, 0, 3);
      }
    }

    // A fragments: lane<16 -> M=l16, K 0-7 & 16-23; lane>=16 -> K 8-15 & 24-31
    Frag af[4], bfg[2];
    const int akb = hi ? 8 : 0;
    #pragma unroll
    for (int ms = 0; ms < 4; ++ms) {
      const int row = wm * 64 + ms * 16 + l16;
      af[ms].u[0] = *(const u32x4*)&smem[SA_IDX(buf, row * 32 + akb)];
      af[ms].u[1] = *(const u32x4*)&smem[SA_IDX(buf, row * 32 + akb + 16)];
    }
    // B fragments: lane = N column, lane<16 -> K 0-7 & 16-23; lane>=16 -> K 8-15 & 24-31
    const int bkb = hi ? 8 : 0;
    #pragma unroll
    for (int ns = 0; ns < 2; ++ns) {
      const int nrow = wn * 32 + ns * 16 + l16;
      bfg[ns].u[0] = *(const u32x4*)&smem[SB_IDX(buf, nrow * 32 + bkb)];
      bfg[ns].u[1] = *(const u32x4*)&smem[SB_IDX(buf, nrow * 32 + bkb + 16)];
    }

    #pragma unroll
    for (int ms = 0; ms < 4; ++ms)
      #pragma unroll
      for (int ns = 0; ns < 2; ++ns)
        acc[ms][ns] = __builtin_amdgcn_wmma_f32_16x16x32_bf16(
            false, af[ms].v, false, bfg[ns].v, (short)0, acc[ms][ns], false, false);

    if (more) {
      async_wait0();       // my async stores to buf^1 landed
      __syncthreads();     // everyone's landed; everyone done reading buf
    }
  }

  // Epilogue: C/D layout: lane<16 -> M r+0..7, lane>=16 -> M r+8..15; N = l16
  #pragma unroll
  for (int ms = 0; ms < 4; ++ms) {
    #pragma unroll
    for (int ns = 0; ns < 2; ++ns) {
      const long n  = blockCol + wn * 32 + ns * 16 + l16;
      const float bv = bias ? bias[n] : 0.f;
      const long mb = blockRow + wm * 64 + ms * 16 + (hi ? 8 : 0);
      #pragma unroll
      for (int r = 0; r < 8; ++r) {
        float v = acc[ms][ns][r] + bv;
        if (act == 1)      v = v * (1.f / (1.f + expf(-v)));                // SiLU
        else if (act == 2) v = 0.5f * v * (1.f + erff(v * 0.70710678118f)); // exact GELU
        const long idx = (mb + r) * (long)N + n;
        if (res)   v += res[idx];
        if (outF)  outF[idx]  = v;
        if (outBf) outBf[idx] = __float2bfloat16(v);
      }
    }
  }
}

// ---------------------------------------------------------------------------
// Weight convert + transpose: fp32 [K][N] -> bf16 [N][Kpad] (zero padded)
// ---------------------------------------------------------------------------
__global__ void wconv_kernel(const float* __restrict__ w, bf16* __restrict__ out,
                             int K, int N, int Kpad) {
  int idx = blockIdx.x * 256 + threadIdx.x;
  if (idx >= N * Kpad) return;
  int n = idx / Kpad, k = idx % Kpad;
  float v = (k < K) ? w[(long)k * N + n] : 0.f;
  out[idx] = __float2bfloat16(v);
}

// Sinusoidal timestep embedding -> bf16 (B, 512): [cos(t*f) | sin(t*f)]
__global__ void temb_kernel(const int* __restrict__ ts, bf16* __restrict__ out) {
  int idx = blockIdx.x * 256 + threadIdx.x;   // < 2048*512
  int b = idx >> 9, i = idx & 511;
  float t = (float)ts[b];
  int j = i & 255;
  float f = expf(-9.210340371976184f * (float)j * (1.0f / 256.0f));
  float a = t * f;
  out[idx] = __float2bfloat16(i < 256 ? cosf(a) : sinf(a));
}

// x (B,T,66) fp32 -> bf16 (B*T, 96) zero padded
__global__ void xpad_kernel(const float* __restrict__ x, bf16* __restrict__ out) {
  int idx = blockIdx.x * 256 + threadIdx.x;
  if (idx >= XROWS * 96) return;
  int row = idx / 96, k = idx % 96;
  float v = (k < DIN_) ? x[(long)row * DIN_ + k] : 0.f;
  out[idx] = __float2bfloat16(v);
}

// h[b,s,:] = (s==0 ? emb[b] : jointOut[b, s-1]) + seq_emb[s]
__global__ void assemble_kernel(const float* __restrict__ jointOut,
                                const float* __restrict__ emb,
                                const float* __restrict__ seq,
                                float* __restrict__ h) {
  int idx = blockIdx.x * 256 + threadIdx.x;   // < ROWS*512
  int row = idx >> 9, d = idx & 511;
  int b = row / S_, s = row % S_;
  float v = (s == 0) ? emb[(long)b * D_ + d]
                     : jointOut[((long)b * T_ + (s - 1)) * D_ + d];
  h[idx] = v + seq[(long)s * D_ + d];
}

// cat = [bf16(h) | skip_bf]  (B*S, 1024)
__global__ void cat_kernel(const float* __restrict__ h, const bf16* __restrict__ skip,
                           bf16* __restrict__ out) {
  int idx = blockIdx.x * 256 + threadIdx.x;   // < ROWS*1024
  int row = idx >> 10, c = idx & 1023;
  out[idx] = (c < 512) ? __float2bfloat16(h[(long)row * D_ + c])
                       : skip[(long)row * D_ + (c - 512)];
}

// SE gate over tokens: one wave per batch element. x *= (1 + sigmoid(relu(s@W1)@W2))
__global__ void se_kernel(float* __restrict__ h, const float* __restrict__ W1,
                          const float* __restrict__ W2) {
  int b = blockIdx.x;
  int lane = threadIdx.x;
  __shared__ float s[S_];
  __shared__ float r[8];
  if (lane < S_) {
    const float* row = h + ((long)b * S_ + lane) * D_;
    float sum = 0.f;
    for (int d = 0; d < D_; ++d) sum += row[d];
    s[lane] = sum * (1.0f / (float)D_);
  }
  __syncthreads();
  if (lane < 5) {
    float a = 0.f;
    for (int t = 0; t < S_; ++t) a += s[t] * W1[t * 5 + lane];
    r[lane] = fmaxf(a, 0.f);
  }
  __syncthreads();
  if (lane < S_) {
    float a = 0.f;
    for (int j = 0; j < 5; ++j) a += r[j] * W2[j * S_ + lane];
    float m = 1.f + 1.f / (1.f + expf(-a));
    float* row = h + ((long)b * S_ + lane) * D_;
    for (int d = 0; d < D_; ++d) row[d] *= m;
  }
}

// LayerNorm: one wave per row (D=512, 16 elems/lane) -> bf16 out
__global__ void ln_kernel(const float* __restrict__ x, const float* __restrict__ g,
                          const float* __restrict__ b, bf16* __restrict__ out,
                          int rows) {
  int row = blockIdx.x * 8 + (threadIdx.x >> 5);
  if (row >= rows) return;
  int lane = threadIdx.x & 31;
  const float* xr = x + (long)row * D_;
  float v[16];
  float s = 0.f;
  #pragma unroll
  for (int i = 0; i < 16; ++i) { v[i] = xr[lane + 32 * i]; s += v[i]; }
  #pragma unroll
  for (int o = 16; o > 0; o >>= 1) s += __shfl_xor(s, o);
  float mean = s * (1.0f / (float)D_);
  float var = 0.f;
  #pragma unroll
  for (int i = 0; i < 16; ++i) { float d = v[i] - mean; var += d * d; }
  #pragma unroll
  for (int o = 16; o > 0; o >>= 1) var += __shfl_xor(var, o);
  var *= (1.0f / (float)D_);
  float inv = rsqrtf(var + 1e-5f);
  #pragma unroll
  for (int i = 0; i < 16; ++i) {
    int c = lane + 32 * i;
    out[(long)row * D_ + c] = __float2bfloat16((v[i] - mean) * inv * g[c] + b[c]);
  }
}

// Attention: one wave per (b, head). qkv bf16 (B,S,[q|k|v]=1536). h += y in-place.
__global__ void attn_kernel(const bf16* __restrict__ qkv, float* __restrict__ h) {
  int bh = blockIdx.x;
  int b = bh >> 3, hd = bh & 7;
  int lane = threadIdx.x;
  __shared__ float kk[S_ * DH_];
  __shared__ float vv[S_ * DH_];
  const bf16* base = qkv + (long)b * S_ * 1536;
  for (int i = lane; i < S_ * DH_; i += 32) {
    int t = i >> 6, d = i & 63;
    kk[i] = __bfloat162float(base[t * 1536 + 512 + hd * DH_ + d]);
    vv[i] = __bfloat162float(base[t * 1536 + 1024 + hd * DH_ + d]);
  }
  __syncthreads();
  if (lane < S_) {
    float q[DH_];
    const bf16* qp = base + lane * 1536 + hd * DH_;
    #pragma unroll
    for (int d = 0; d < DH_; ++d) q[d] = __bfloat162float(qp[d]);
    float sc[S_];
    float mx = -1e30f;
    for (int m = 0; m < S_; ++m) {
      float a = 0.f;
      #pragma unroll
      for (int d = 0; d < DH_; ++d) a += q[d] * kk[m * DH_ + d];
      a *= 0.125f;                    // 1/sqrt(64)
      sc[m] = a;
      mx = fmaxf(mx, a);
    }
    float den = 0.f;
    for (int m = 0; m < S_; ++m) { sc[m] = expf(sc[m] - mx); den += sc[m]; }
    float inv = 1.f / den;
    float y[DH_];
    #pragma unroll
    for (int d = 0; d < DH_; ++d) y[d] = 0.f;
    for (int m = 0; m < S_; ++m) {
      float w = sc[m] * inv;
      #pragma unroll
      for (int d = 0; d < DH_; ++d) y[d] += w * vv[m * DH_ + d];
    }
    float* hr = h + ((long)b * S_ + lane) * D_ + hd * DH_;
    #pragma unroll
    for (int d = 0; d < DH_; ++d) hr[d] += y[d];
  }
}

// Output projection: h[:,1:,:](512) @ out_W(512,66) + out_b -> d_out (B,T,66). One wave/row.
__global__ void outproj_kernel(const float* __restrict__ h, const float* __restrict__ W,
                               const float* __restrict__ bb, float* __restrict__ out) {
  int row = blockIdx.x * 8 + (threadIdx.x >> 5);   // < XROWS
  int lane = threadIdx.x & 31;
  int b = row / T_, t = row % T_;
  const float* hr = h + ((long)b * S_ + t + 1) * D_;
  for (int n = lane; n < DIN_; n += 32) {
    float a = bb[n];
    for (int k = 0; k < D_; ++k) a += hr[k] * W[(long)k * DIN_ + n];
    out[(long)row * DIN_ + n] = a;
  }
}

// ---------------------------------------------------------------------------
// Host orchestration
// ---------------------------------------------------------------------------
extern "C" void kernel_launch(void* const* d_in, const int* in_sizes, int n_in,
                              void* d_out, int out_size, void* d_ws, size_t ws_size,
                              hipStream_t stream) {
  (void)in_sizes; (void)n_in; (void)out_size; (void)ws_size;

  // ---- unpack inputs in setup_inputs() flattening order ----
  int ii = 0;
  auto nf = [&]() { return (const float*)d_in[ii++]; };
  const float* x       = nf();
  const int*   tsteps  = (const int*)d_in[ii++];
  const float* seq_emb = nf();
  const float* jointW  = nf(); const float* jointB = nf();
  const float* t1W = nf();     const float* t1B = nf();
  const float* t2W = nf();     const float* t2B = nf();

  struct Blk {
    const float *sag, *sab, *Wq, *Wk, *Wv, *ffg, *ffb,
                *f1W, *f1b, *f2W, *f2b, *se1, *se2, *skW, *skb;
  };
  Blk blk[9];
  auto readBlk = [&](bool skip) {
    Blk p{};
    p.sag = nf(); p.sab = nf(); p.Wq = nf(); p.Wk = nf(); p.Wv = nf();
    p.ffg = nf(); p.ffb = nf(); p.f1W = nf(); p.f1b = nf();
    p.f2W = nf(); p.f2b = nf(); p.se1 = nf(); p.se2 = nf();
    if (skip) { p.skW = nf(); p.skb = nf(); }
    return p;
  };
  for (int j = 0; j < 4; ++j) blk[j] = readBlk(false);
  blk[4] = readBlk(false);
  for (int j = 5; j < 9; ++j) blk[j] = readBlk(true);
  const float* outW = nf();
  const float* outB = nf();

  // ---- workspace carve-up ----
  size_t off = 0;
  auto take = [&](size_t bytes) {
    size_t o = off;
    off += (bytes + 255) & ~(size_t)255;
    return (char*)d_ws + o;
  };
  float* hBuf   = (float*)take((size_t)ROWS * D_ * 4);
  bf16*  xnBf   = (bf16*) take((size_t)ROWS * D_ * 2);
  bf16*  qkvBf  = (bf16*) take((size_t)ROWS * 1536 * 2);
  bf16*  midBf  = (bf16*) take((size_t)ROWS * 1024 * 2);   // also used as cat buffer
  bf16*  skipBf[4];
  for (int j = 0; j < 4; ++j) skipBf[j] = (bf16*)take((size_t)ROWS * D_ * 2);
  float* embF   = (float*)take((size_t)B_ * D_ * 4);
  bf16*  tembBf = (bf16*) take((size_t)B_ * D_ * 2);
  bf16*  t1Bf   = (bf16*) take((size_t)B_ * D_ * 2);
  bf16*  xPadBf = (bf16*) take((size_t)XROWS * 96 * 2);
  bf16*  t1Wbf  = (bf16*) take((size_t)512 * 512 * 2);
  bf16*  t2Wbf  = (bf16*) take((size_t)512 * 512 * 2);
  bf16*  jWbf   = (bf16*) take((size_t)512 * 96 * 2);
  bf16*  qkvW[9]; bf16* f1W[9]; bf16* f2W[9]; bf16* skW[4];
  for (int j = 0; j < 9; ++j) {
    qkvW[j] = (bf16*)take((size_t)1536 * 512 * 2);
    f1W[j]  = (bf16*)take((size_t)1024 * 512 * 2);
    f2W[j]  = (bf16*)take((size_t)512 * 1024 * 2);
  }
  for (int j = 0; j < 4; ++j) skW[j] = (bf16*)take((size_t)512 * 1024 * 2);
  float* jointOut = (float*)qkvBf;   // reuse: qkv region free until first attention

  // ---- convert/transpose all GEMM weights to bf16 [N][Kpad] ----
  auto wconvL = [&](const float* w, bf16* o, int K, int N, int Kpad) {
    int total = N * Kpad;
    wconv_kernel<<<(total + 255) / 256, 256, 0, stream>>>(w, o, K, N, Kpad);
  };
  wconvL(t1W, t1Wbf, 512, 512, 512);
  wconvL(t2W, t2Wbf, 512, 512, 512);
  wconvL(jointW, jWbf, 66, 512, 96);
  for (int j = 0; j < 9; ++j) {
    wconvL(blk[j].Wq, qkvW[j],              512, 512, 512);
    wconvL(blk[j].Wk, qkvW[j] + 512 * 512,  512, 512, 512);
    wconvL(blk[j].Wv, qkvW[j] + 1024 * 512, 512, 512, 512);
    wconvL(blk[j].f1W, f1W[j], 512, 1024, 512);
    wconvL(blk[j].f2W, f2W[j], 1024, 512, 1024);
    if (j >= 5) wconvL(blk[j].skW, skW[j - 5], 1024, 512, 1024);
  }

  auto gemm = [&](const bf16* A, const bf16* Bw, const float* bias, const float* res,
                  float* oF, bf16* oB, int M, int N, int K, int act) {
    dim3 g(M / 128, N / 128);
    gemm_bf16_kernel<<<g, 256, 0, stream>>>(A, Bw, bias, res, oF, oB, M, N, K, act);
  };

  // ---- timestep embedding MLP ----
  temb_kernel<<<(B_ * D_) / 256, 256, 0, stream>>>(tsteps, tembBf);
  gemm(tembBf, t1Wbf, t1B, nullptr, nullptr, t1Bf, B_, 512, 512, 1);   // SiLU
  gemm(t1Bf,   t2Wbf, t2B, nullptr, embF,    nullptr, B_, 512, 512, 0);

  // ---- joint projection + sequence assembly ----
  xpad_kernel<<<(XROWS * 96 + 255) / 256, 256, 0, stream>>>(x, xPadBf);
  gemm(xPadBf, jWbf, jointB, nullptr, jointOut, nullptr, XROWS, 512, 96, 0);
  assemble_kernel<<<(ROWS * D_) / 256, 256, 0, stream>>>(jointOut, embF, seq_emb, hBuf);

  // ---- 9 transformer blocks (4 in, 1 mid, 4 out-with-skip) ----
  for (int j = 0; j < 9; ++j) {
    const Blk& p = blk[j];
    if (j >= 5) {  // skip fusion: x = concat([x, skip]) @ skW + skb
      bf16* cat = midBf;
      cat_kernel<<<(ROWS * 1024) / 256, 256, 0, stream>>>(hBuf, skipBf[8 - j], cat);
      gemm(cat, skW[j - 5], p.skb, nullptr, hBuf, nullptr, ROWS, 512, 1024, 0);
    }
    se_kernel<<<B_, 32, 0, stream>>>(hBuf, p.se1, p.se2);
    ln_kernel<<<ROWS / 8, 256, 0, stream>>>(hBuf, p.sag, p.sab, xnBf, ROWS);
    gemm(xnBf, qkvW[j], nullptr, nullptr, nullptr, qkvBf, ROWS, 1536, 512, 0);
    attn_kernel<<<B_ * H_, 32, 0, stream>>>(qkvBf, hBuf);
    ln_kernel<<<ROWS / 8, 256, 0, stream>>>(hBuf, p.ffg, p.ffb, xnBf, ROWS);
    gemm(xnBf, f1W[j], p.f1b, nullptr, nullptr, midBf, ROWS, 1024, 512, 2);   // GELU
    gemm(midBf, f2W[j], p.f2b, hBuf, hBuf, (j < 4) ? skipBf[j] : nullptr,
         ROWS, 512, 1024, 0);   // residual add; in-blocks also capture skip as bf16
  }

  // ---- output projection ----
  outproj_kernel<<<XROWS / 8, 256, 0, stream>>>(hBuf, outW, outB, (float*)d_out);
}
